// LIF_75591424409727
// MI455X (gfx1250) — matrix operand order repaired and checked by reading, not compiled, not run on hardware
//
#include <hip/hip_runtime.h>

// ---------------------------------------------------------------------------
// Fused LIF forward: x = inputs @ w  (bf16 WMMA, f32 accumulate)  +  LIF scan
// inputs: [64, 1024, 256] f32, w: [256, 256] f32  ->  U: [64, 1024, 256] f32
// ---------------------------------------------------------------------------

typedef __attribute__((ext_vector_type(16))) __bf16 v16bf;
typedef __attribute__((ext_vector_type(2)))  __bf16 v2bf;
typedef __attribute__((ext_vector_type(8)))  float  v8f;

constexpr int kB   = 64;
constexpr int kT   = 1024;
constexpr int kIN  = 256;   // K
constexpr int kOUT = 256;   // N
constexpr int kHalf    = 128;  // neurons handled per block
constexpr int kThreads = 128;  // 4 waves (wave32)
constexpr int kChunk   = 32;   // timesteps per chunk = two 16-row M tiles
constexpr int A_S = 264;       // a_lds row stride in bf16 (spread banks)
constexpr int X_S = 132;       // x_lds row stride in f32

// exp(-DT/TAU_SYN), exp(-DT/TAU_MEM), 1-exp(-DT/TAU_MEM)
#define DCY_SYN 0.8187307530779818f
#define DCY_MEM 0.9048374180359595f
#define SCL_MEM 0.0951625819640405f

__device__ __forceinline__ v2bf cvt2bf(float a, float b) {
#if __has_builtin(__builtin_amdgcn_cvt_pk_bf16_f32)
  return __builtin_amdgcn_cvt_pk_bf16_f32(a, b);
#else
  v2bf r; r[0] = (__bf16)a; r[1] = (__bf16)b; return r;
#endif
}

__device__ __forceinline__ v16bf load_afrag(const __bf16* base) {
  // A 16x32 bf16 layout: lanes 0-15 (row M=lane): elems 0-7 = K 0..7,
  // elems 8-15 = K 16..23; lanes 16-31 (row M=lane-16): K 8..15 / 24..31.
  // `base` = &row[kb*32 + lhi*8].
  v16bf a;
  #pragma unroll
  for (int e = 0; e < 8; ++e) a[e] = base[e];
  #pragma unroll
  for (int e = 0; e < 8; ++e) a[8 + e] = base[16 + e];
  return a;
}

__global__ __launch_bounds__(kThreads, 1)
void lif_fused_kernel(const float* __restrict__ inp,
                      const float* __restrict__ w,
                      float* __restrict__ out)
{
  __shared__ __bf16 a_lds[kChunk * A_S];   // 32 x 256 input tile (bf16)
  __shared__ float  x_lds[kChunk * X_S];   // 32 x 128 GEMM result tile (f32)

  const int b     = blockIdx.x;
  const int nbase = blockIdx.y * kHalf;
  const int tid   = threadIdx.x;
  const int lane  = tid & 31;
  const int wave  = tid >> 5;
  const int lrow  = lane & 15;   // A: row M; B: column within N-tile
  const int lhi   = lane >> 4;   // lane half select

  // U[:, 0, :] = 0 (initial stored state)
  out[(size_t)b * kT * kOUT + nbase + tid] = 0.0f;

  // ---- Preload B (weight) fragments into registers: 2 N-tiles x 8 K-blocks.
  // B 32x16 bf16 layout: lanes 0-15 hold K=0..7 / 16..23 (elems 0..15) for column
  // N=lane; lanes 16-31 hold K=8..15 / 24..31 for column N=lane-16.
  v16bf bfrag[2][8];
  #pragma unroll
  for (int i = 0; i < 2; ++i) {
    const int n = nbase + wave * 32 + i * 16 + lrow;
    #pragma unroll
    for (int kb = 0; kb < 8; ++kb) {
      const int k0 = kb * 32 + lhi * 8;
      #pragma unroll
      for (int e = 0; e < 16; ++e)
        bfrag[i][kb][e] = (__bf16)w[(size_t)(k0 + e + (e >= 8 ? 8 : 0)) * kOUT + n];
    }
  }

  float syn = 0.0f, mem = 0.0f;
  const float* inb = inp + (size_t)b * kT * kIN;
  float* outp = out + (size_t)b * kT * kOUT + nbase + tid;

  for (int c = 0; c < kT / kChunk; ++c) {
    // ---- Stage 32x256 input tile to LDS as bf16 (tile is contiguous) ----
    const float* src = inb + (size_t)c * kChunk * kIN;
    for (int idx = tid; idx < (kChunk * kIN) / 4; idx += kThreads) {
      const float4 v = ((const float4*)src)[idx];
      if (c + 1 < kT / kChunk)
        __builtin_prefetch(src + kChunk * kIN + idx * 4, 0, 1);  // warm next tile
      const int flat = idx * 4;
      const int r = flat >> 8, col = flat & 255;
      v2bf* d = (v2bf*)&a_lds[r * A_S + col];
      d[0] = cvt2bf(v.x, v.y);
      d[1] = cvt2bf(v.z, v.w);
    }
    __syncthreads();

    // ---- WMMA: this wave computes x[32 x 32] = A(32x256) * B(256x32) ----
    v8f acc00 = {}, acc01 = {}, acc10 = {}, acc11 = {};
    #pragma unroll
    for (int kb = 0; kb < 8; ++kb) {
      const __bf16* base0 = &a_lds[lrow * A_S + kb * 32 + lhi * 8];
      const v16bf a0 = load_afrag(base0);                 // M rows 0..15
      const v16bf a1 = load_afrag(base0 + 16 * A_S);      // M rows 16..31
      acc00 = __builtin_amdgcn_wmma_f32_16x16x32_bf16(
                  false, a0, false, bfrag[0][kb], (short)0, acc00, false, false);
      acc01 = __builtin_amdgcn_wmma_f32_16x16x32_bf16(
                  false, a0, false, bfrag[1][kb], (short)0, acc01, false, false);
      acc10 = __builtin_amdgcn_wmma_f32_16x16x32_bf16(
                  false, a1, false, bfrag[0][kb], (short)0, acc10, false, false);
      acc11 = __builtin_amdgcn_wmma_f32_16x16x32_bf16(
                  false, a1, false, bfrag[1][kb], (short)0, acc11, false, false);
    }

    // ---- Scatter C/D to LDS: VGPR v -> M = v + 8*lhi, N = tile_n0 + lrow ----
    {
      const int nloc = wave * 32 + lrow;
      #pragma unroll
      for (int v = 0; v < 8; ++v) {
        const int m = v + 8 * lhi;
        x_lds[m * X_S + nloc]             = acc00[v];
        x_lds[m * X_S + nloc + 16]        = acc01[v];
        x_lds[(m + 16) * X_S + nloc]      = acc10[v];
        x_lds[(m + 16) * X_S + nloc + 16] = acc11[v];
      }
    }
    __syncthreads();

    // ---- LIF scan: thread `tid` owns one neuron, 32 sequential steps ----
    #pragma unroll
    for (int tt = 0; tt < kChunk; ++tt) {
      const int t = c * kChunk + tt;
      if (t < kT - 1) {                       // loop runs t = 0 .. T-2
        const float xt      = x_lds[tt * X_S + tid];
        const float new_syn = DCY_SYN * syn + xt;
        float       new_mem = fmaf(SCL_MEM, syn, DCY_MEM * mem);  // OLD syn
        new_mem = (mem > 1.0f) ? 0.0f : new_mem;                  // spike->reset
        outp[(size_t)(t + 1) * kOUT] = new_mem;                   // U[:, t+1, :]
        syn = new_syn; mem = new_mem;
      }
    }
    __syncthreads();
  }
}

extern "C" void kernel_launch(void* const* d_in, const int* in_sizes, int n_in,
                              void* d_out, int out_size, void* d_ws, size_t ws_size,
                              hipStream_t stream) {
  (void)in_sizes; (void)n_in; (void)out_size; (void)d_ws; (void)ws_size;
  const float* inp = (const float*)d_in[0];   // [64,1024,256] f32
  const float* w   = (const float*)d_in[1];   // [256,256] f32
  float* out = (float*)d_out;                 // [64,1024,256] f32

  dim3 grid(kB, kOUT / kHalf);                // 64 x 2 = 128 workgroups
  hipLaunchKernelGGL(lif_fused_kernel, grid, dim3(kThreads), 0, stream,
                     inp, w, out);
}
